// CWCT_1417339208094
// MI455X (gfx1250) — compile-verified, matrix operand order checked by reading, not run on hardware
//
#include <hip/hip_runtime.h>
#include <stdint.h>

// ---------------------------------------------------------------------------
// Segmented WCT for MI455X (gfx1250, wave32).
// Fast path: pre-packed bf16 operands + precomputed per-label AND-masks so the
// WMMA inner loops are ~1 v_and per operand register per label (XDL/BW bound,
// not VALU bound). Falls back to direct-f32 kernels if ws is too small.
// ---------------------------------------------------------------------------

#define NLAB 8
#define CCH  256
#define NPIX (512 * 512)
#define NPAIR (NPIX / 2)          // 131072 bf16-pairs per channel row

typedef __attribute__((ext_vector_type(16))) __bf16 v16bf;
typedef __attribute__((ext_vector_type(8)))  float  v8f;

// ---- workspace layout (units: float/u32 slots from ws base) ----------------
#define OFF_SUM_C 0
#define OFF_SUM_S 2048
#define OFF_CNT_C 4096
#define OFF_CNT_S 4104
#define OFF_MU_C  4112
#define OFF_MU_S  6160
#define OFF_GUIDE 8208
#define OFF_COV_C 8224
#define OFF_COV_S (8224 + 524288)
#define OFF_INVL  (8224 + 2 * 524288)
#define OFF_T     (8224 + 3 * 524288)
#define BASE_END  (8224 + 4 * 524288)              // 2105376
#define ZERO_FLOATS (8224 + 2 * 524288)

// fast-path extras (u32 slots)
#define OFF_TBF   BASE_END                          // 8*16*8*32*8   = 262144
#define OFF_MSK_C (OFF_TBF + 262144)                // 8*NPAIR       = 1048576
#define OFF_MSK_S (OFF_MSK_C + 1048576)
#define OFF_XBF_C (OFF_MSK_S + 1048576)             // 256*NPAIR     = 33554432
#define OFF_XBF_S (OFF_XBF_C + 33554432)
#define OFF_XTC   (OFF_XBF_S + 33554432)            // NPIX*128      = 33554432
#define FAST_END  (OFF_XTC + 33554432)
#define FAST_BYTES ((size_t)FAST_END * 4)

__device__ __forceinline__ uint32_t pack2bf(float a, float b) {
    uint32_t ua = __float_as_uint(a);
    uint32_t ub = __float_as_uint(b);
    ua += 0x7FFFu + ((ua >> 16) & 1u);              // round-to-nearest-even
    ub += 0x7FFFu + ((ub >> 16) & 1u);
    return (ua >> 16) | (ub & 0xFFFF0000u);
}

__device__ __forceinline__ v16bf as_v16bf(const uint32_t* u) {
    union { uint32_t u[8]; v16bf v; } c;
#pragma unroll
    for (int i = 0; i < 8; ++i) c.u[i] = u[i];
    return c.v;
}

// ================================================================ shared path
__global__ void k_zero(float* ws, int n) {
    int i = blockIdx.x * blockDim.x + threadIdx.x;
    int stride = gridDim.x * blockDim.x;
    for (; i < n; i += stride) ws[i] = 0.f;
}

__global__ void k_hist(const int* __restrict__ seg, float* __restrict__ cnt) {
    __shared__ float h[NLAB];
    if (threadIdx.x < NLAB) h[threadIdx.x] = 0.f;
    __syncthreads();
    int i = blockIdx.x * blockDim.x + threadIdx.x;
    int stride = gridDim.x * blockDim.x;
    for (; i < NPIX; i += stride) atomicAdd(&h[seg[i]], 1.f);
    __syncthreads();
    if (threadIdx.x < NLAB) atomicAdd(&cnt[threadIdx.x], h[threadIdx.x]);
}

__global__ void k_sums(const float* __restrict__ x, const int* __restrict__ seg,
                       float* __restrict__ sum) {
    const int c = blockIdx.x;
    const int per = NPIX / gridDim.y;
    const float* xr = x + (size_t)c * NPIX + (size_t)blockIdx.y * per;
    const int* sr = seg + blockIdx.y * per;
    float loc[NLAB];
#pragma unroll
    for (int l = 0; l < NLAB; ++l) loc[l] = 0.f;
    for (int i = threadIdx.x; i < per; i += blockDim.x) {
        int lab = sr[i];
        float v = xr[i];
#pragma unroll
        for (int l = 0; l < NLAB; ++l) loc[l] += (lab == l) ? v : 0.f;
    }
    __shared__ float h[NLAB];
    if (threadIdx.x < NLAB) h[threadIdx.x] = 0.f;
    __syncthreads();
#pragma unroll
    for (int l = 0; l < NLAB; ++l)
        if (loc[l] != 0.f) atomicAdd(&h[l], loc[l]);
    __syncthreads();
    if (threadIdx.x < NLAB) atomicAdd(&sum[threadIdx.x * CCH + c], h[threadIdx.x]);
}

__global__ void k_means(float* ws) {
    int l = blockIdx.x, c = threadIdx.x;
    float a = ws[OFF_CNT_C + l], b = ws[OFF_CNT_S + l];
    ws[OFF_MU_C + l * CCH + c] = ws[OFF_SUM_C + l * CCH + c] / fmaxf(a, 1.f);
    ws[OFF_MU_S + l * CCH + c] = ws[OFF_SUM_S + l * CCH + c] / fmaxf(b, 1.f);
    if (c == 0) {
        bool g = (a > 10.f) && (b > 10.f) && (a < 100.f * b) && (b < 100.f * a);
        ws[OFF_GUIDE + l] = g ? 1.f : 0.f;
    }
}

__global__ void k_covfin(float* ws) {
    int l = blockIdx.x >> 8;
    int i = blockIdx.x & 255;
    int j = threadIdx.x;
    int isStyle = blockIdx.z;
    float* cov = ws + (isStyle ? OFF_COV_S : OFF_COV_C);
    float cnt = ws[(isStyle ? OFF_CNT_S : OFF_CNT_C) + l];
    const float* mu = ws + (isStyle ? OFF_MU_S : OFF_MU_C) + l * CCH;
    float g = ws[OFF_GUIDE + l];
    size_t idx = (size_t)l * 65536 + i * 256 + j;
    float v;
    if (g > 0.5f) v = (cov[idx] - cnt * mu[i] * mu[j]) / fmaxf(cnt - 1.f, 1.f);
    else          v = (i == j) ? 1.f : 0.f;
    cov[idx] = v;
}

__global__ void k_chol(float* ws) {
    int m = blockIdx.x;
    float* A = ws + ((m < 8) ? (OFF_COV_C + m * 65536)
                             : (OFF_COV_S + (m - 8) * 65536));
    int i = threadIdx.x;
    for (int k = 0; k < CCH; ++k) {
        if (i == k) A[k * CCH + k] = sqrtf(fmaxf(A[k * CCH + k], 1e-20f));
        __syncthreads();
        float dk = A[k * CCH + k];
        if (i > k) A[i * CCH + k] /= dk;
        __syncthreads();
        if (i > k) {
            float lik = A[i * CCH + k];
            for (int j = k + 1; j <= i; ++j)
                A[i * CCH + j] -= lik * A[j * CCH + k];
        }
        __syncthreads();
    }
}

__global__ void k_trinv(float* ws) {
    int l = blockIdx.x, j = threadIdx.x;
    const float* L = ws + OFF_COV_C + l * 65536;
    float* Z = ws + OFF_INVL + l * 65536;
    float z[CCH];
    for (int i = 0; i < j; ++i) z[i] = 0.f;
    z[j] = 1.f / L[j * CCH + j];
    for (int i = j + 1; i < CCH; ++i) {
        float s = 0.f;
        for (int k = j; k < i; ++k) s += L[i * CCH + k] * z[k];
        z[i] = -s / L[i * CCH + i];
    }
    for (int i = 0; i < CCH; ++i) Z[i * CCH + j] = z[i];
}

__global__ void k_formT(float* ws) {
    int l = blockIdx.x, i = blockIdx.y, j = threadIdx.x;
    const float* Ls = ws + OFF_COV_S + l * 65536;
    const float* Z  = ws + OFF_INVL  + l * 65536;
    float s = 0.f;
    if (j <= i)
        for (int k = j; k <= i; ++k) s += Ls[i * CCH + k] * Z[k * CCH + j];
    ws[OFF_T + l * 65536 + i * CCH + j] = s;
}

// ================================================================ fast preps
// x (flat [C*N] f32) -> packed bf16 pairs
__global__ void k_packx(const float* __restrict__ x, uint32_t* __restrict__ xbf) {
    int i = blockIdx.x * blockDim.x + threadIdx.x;
    int stride = gridDim.x * blockDim.x;
    const int n = CCH * NPAIR;
    for (; i < n; i += stride)
        xbf[i] = pack2bf(x[2 * (size_t)i], x[2 * (size_t)i + 1]);
}

// per-label pair masks: 0xFFFF per matching bf16 half
__global__ void k_masks(const int* __restrict__ seg, uint32_t* __restrict__ msk) {
    int i = blockIdx.x * blockDim.x + threadIdx.x;
    int stride = gridDim.x * blockDim.x;
    const int n = NLAB * NPAIR;
    for (; i < n; i += stride) {
        int l = i >> 17, p = i & (NPAIR - 1);
        int l0 = seg[2 * p], l1 = seg[2 * p + 1];
        msk[i] = (l0 == l ? 0x0000FFFFu : 0u) | (l1 == l ? 0xFFFF0000u : 0u);
    }
}

// T (f32) -> WMMA-tiled bf16: Tbf[((l*16+itile)*8+ch)*256 + lane*8 + v]
__global__ void k_packT(const float* __restrict__ Tf, uint32_t* __restrict__ Tbf) {
    int i = blockIdx.x * blockDim.x + threadIdx.x;
    int stride = gridDim.x * blockDim.x;
    for (; i < 262144; i += stride) {
        int v = i & 7, lane = (i >> 3) & 31;
        int rest = i >> 8;
        int ch = rest & 7, itile = (rest >> 3) & 15, l = rest >> 7;
        int row = itile * 16 + (lane & 15);
        int half = lane >> 4;
        int k = ch * 32 + ((v >> 2) << 4) + (half << 3) + ((v & 3) << 1);
        const float* Tl = Tf + (size_t)l * 65536 + row * CCH;
        Tbf[i] = pack2bf(Tl[k], Tl[k + 1]);
    }
}

// centered + transposed content: xtc[px*128 + pair] = bf16(x - mu_c[lab(px)])
__global__ void __launch_bounds__(256)
k_xtc(const float* __restrict__ x, const int* __restrict__ seg,
      const float* __restrict__ ws, uint32_t* __restrict__ xtc) {
    __shared__ float mu[NLAB * CCH];
    for (int i = threadIdx.x; i < NLAB * CCH; i += blockDim.x)
        mu[i] = ws[OFF_MU_C + i];
    __syncthreads();
    const int px = blockIdx.x * blockDim.x + threadIdx.x;
    const float* mrow = &mu[seg[px] * CCH];
    uint32_t* o = xtc + (size_t)px * 128;
#pragma unroll 4
    for (int p = 0; p < 128; ++p) {
        int c = 2 * p;
        float v0 = x[(size_t)c * NPIX + px] - mrow[c];
        float v1 = x[(size_t)(c + 1) * NPIX + px] - mrow[c + 1];
        o[p] = pack2bf(v0, v1);
    }
}

// ================================================================ cov (fast)
// One wave per (i-tile, j-tile, splitK, content/style): b128 operand loads,
// per-label masking = 8 v_and per operand, 8 WMMAs per 32-pixel chunk.
__global__ void __launch_bounds__(32)
k_cov_fast(float* __restrict__ ws) {
    const int pair  = blockIdx.x;
    const int itile = pair >> 4, jtile = pair & 15;
    const int isStyle = blockIdx.z;
    const uint32_t* ws32 = (const uint32_t*)ws;
    const uint32_t* xbf = ws32 + (isStyle ? OFF_XBF_S : OFF_XBF_C);
    const uint32_t* msk = ws32 + (isStyle ? OFF_MSK_S : OFF_MSK_C);
    float* cov = ws + (isStyle ? OFF_COV_S : OFF_COV_C);

    const int lane = threadIdx.x & 31;
    const int half = lane >> 4;
    const uint32_t* xa = xbf + (size_t)(itile * 16 + (lane & 15)) * NPAIR;
    const uint32_t* xb = xbf + (size_t)(jtile * 16 + (lane & 15)) * NPAIR;

    const v8f vzero = {0.f, 0.f, 0.f, 0.f, 0.f, 0.f, 0.f, 0.f};
    v8f acc[NLAB];
#pragma unroll
    for (int l = 0; l < NLAB; ++l) acc[l] = vzero;

    const int kchunk = NPIX / gridDim.y;
    const int k0 = blockIdx.y * kchunk, k1 = k0 + kchunk;
    for (int kk = k0; kk < k1; kk += 32) {
        const int p0 = (kk >> 1) + half * 4;     // first pair of this lane
        __builtin_prefetch(xa + p0 + 1024, 0, 1);   // global_prefetch_b8
        __builtin_prefetch(xb + p0 + 1024, 0, 1);
        uint32_t aU[8], bU[8];
        {
            uint4 a0 = *(const uint4*)(xa + p0);
            uint4 a1 = *(const uint4*)(xa + p0 + 8);
            uint4 b0 = *(const uint4*)(xb + p0);
            uint4 b1 = *(const uint4*)(xb + p0 + 8);
            aU[0] = a0.x; aU[1] = a0.y; aU[2] = a0.z; aU[3] = a0.w;
            aU[4] = a1.x; aU[5] = a1.y; aU[6] = a1.z; aU[7] = a1.w;
            bU[0] = b0.x; bU[1] = b0.y; bU[2] = b0.z; bU[3] = b0.w;
            bU[4] = b1.x; bU[5] = b1.y; bU[6] = b1.z; bU[7] = b1.w;
        }
        v16bf B = as_v16bf(bU);
#pragma unroll
        for (int l = 0; l < NLAB; ++l) {
            const uint32_t* ml = msk + (size_t)l * NPAIR + p0;
            uint4 m0 = *(const uint4*)(ml);
            uint4 m1 = *(const uint4*)(ml + 8);
            uint32_t am[8];
            am[0] = aU[0] & m0.x; am[1] = aU[1] & m0.y;
            am[2] = aU[2] & m0.z; am[3] = aU[3] & m0.w;
            am[4] = aU[4] & m1.x; am[5] = aU[5] & m1.y;
            am[6] = aU[6] & m1.z; am[7] = aU[7] & m1.w;
            acc[l] = __builtin_amdgcn_wmma_f32_16x16x32_bf16(
                false, as_v16bf(am), false, B, (short)0, acc[l], false, false);
        }
    }
#pragma unroll
    for (int l = 0; l < NLAB; ++l) {
#pragma unroll
        for (int r = 0; r < 8; ++r) {
            int row = itile * 16 + r + half * 8;
            int col = jtile * 16 + (lane & 15);
            atomicAdd(&cov[l * 65536 + row * 256 + col], acc[l][r]);
        }
    }
}

// ================================================================ apply (fast)
__global__ void __launch_bounds__(256)
k_apply_fast(const float* __restrict__ x, const int* __restrict__ seg,
             const float* __restrict__ ws, float* __restrict__ out) {
    __shared__ float mus[NLAB * CCH];
    __shared__ float gd[NLAB];
    for (int i = threadIdx.x; i < NLAB * CCH; i += blockDim.x)
        mus[i] = ws[OFF_MU_S + i];
    if (threadIdx.x < NLAB) gd[threadIdx.x] = ws[OFF_GUIDE + threadIdx.x];
    __syncthreads();

    const uint32_t* ws32 = (const uint32_t*)ws;
    const uint32_t* Tbf = ws32 + OFF_TBF;
    const uint32_t* xtc = ws32 + OFF_XTC;

    const int wave  = threadIdx.x >> 5;
    const int lane  = threadIdx.x & 31;
    const int itile = blockIdx.y * 8 + wave;
    const int px0   = blockIdx.x * 16;
    const int n = lane & 15, half = lane >> 4;
    const int px = px0 + n;
    const int labn = seg[px];
    const float gn = gd[labn];
    const uint32_t* bp = xtc + (size_t)px * 128;

    const v8f vzero = {0.f, 0.f, 0.f, 0.f, 0.f, 0.f, 0.f, 0.f};
    v8f acc = vzero;
#pragma unroll
    for (int ch = 0; ch < 8; ++ch) {                 // K chunks of 32 channels
        const int pb = ch * 16 + half * 4;
        uint32_t bU[8];
        {
            uint4 b0 = *(const uint4*)(bp + pb);
            uint4 b1 = *(const uint4*)(bp + pb + 8);
            bU[0] = b0.x; bU[1] = b0.y; bU[2] = b0.z; bU[3] = b0.w;
            bU[4] = b1.x; bU[5] = b1.y; bU[6] = b1.z; bU[7] = b1.w;
        }
#pragma unroll
        for (int l = 0; l < NLAB; ++l) {
            if (__any(labn == l)) {                  // uniform: EXEC stays all-1
                const uint32_t* tb = Tbf + ((size_t)((l * 16 + itile) * 8 + ch)) * 256
                                         + lane * 8;
                uint32_t aU[8], bm[8];
                uint4 a0 = *(const uint4*)(tb);
                uint4 a1 = *(const uint4*)(tb + 4);
                aU[0] = a0.x; aU[1] = a0.y; aU[2] = a0.z; aU[3] = a0.w;
                aU[4] = a1.x; aU[5] = a1.y; aU[6] = a1.z; aU[7] = a1.w;
#pragma unroll
                for (int v = 0; v < 8; ++v) bm[v] = (labn == l) ? bU[v] : 0u;
                acc = __builtin_amdgcn_wmma_f32_16x16x32_bf16(
                    false, as_v16bf(aU), false, as_v16bf(bm),
                    (short)0, acc, false, false);
            }
        }
    }
#pragma unroll
    for (int r = 0; r < 8; ++r) {
        int c = itile * 16 + r + half * 8;
        float val = (gn > 0.5f) ? (acc[r] + mus[labn * CCH + c])
                                : x[(size_t)c * NPIX + px];
        out[(size_t)c * NPIX + px] = val;
    }
}

// ================================================================ slow path
__global__ void __launch_bounds__(32)
k_cov_slow(const float* __restrict__ xc, const float* __restrict__ xs,
           const int* __restrict__ segc, const int* __restrict__ segs,
           float* __restrict__ ws) {
    const int pair  = blockIdx.x;
    const int itile = pair >> 4, jtile = pair & 15;
    const int isStyle = blockIdx.z;
    const float* x   = isStyle ? xs : xc;
    const int*   seg = isStyle ? segs : segc;
    float*       cov = ws + (isStyle ? OFF_COV_S : OFF_COV_C);

    const int lane = threadIdx.x & 31;
    const int half = lane >> 4;
    const float* arow = x + (size_t)(itile * 16 + (lane & 15)) * NPIX;
    const float* brow = x + (size_t)(jtile * 16 + (lane & 15)) * NPIX;

    const v8f vzero = {0.f, 0.f, 0.f, 0.f, 0.f, 0.f, 0.f, 0.f};
    v8f acc[NLAB];
#pragma unroll
    for (int l = 0; l < NLAB; ++l) acc[l] = vzero;

    const int kchunk = NPIX / gridDim.y;
    const int k0 = blockIdx.y * kchunk, k1 = k0 + kchunk;
    for (int kk = k0; kk < k1; kk += 32) {
        uint32_t aU[8], bU[8];
        int labL[8], labH[8];
#pragma unroll
        for (int v = 0; v < 8; ++v) {
            int ke = kk + ((v >> 2) << 4) + (half << 3) + ((v & 3) << 1);
            aU[v] = pack2bf(arow[ke], arow[ke + 1]);
            bU[v] = pack2bf(brow[ke], brow[ke + 1]);
            labL[v] = seg[ke];
            labH[v] = seg[ke + 1];
        }
        v16bf B = as_v16bf(bU);
#pragma unroll
        for (int l = 0; l < NLAB; ++l) {
            uint32_t am[8];
#pragma unroll
            for (int v = 0; v < 8; ++v) {
                uint32_t m = (labL[v] == l ? 0x0000FFFFu : 0u) |
                             (labH[v] == l ? 0xFFFF0000u : 0u);
                am[v] = aU[v] & m;
            }
            acc[l] = __builtin_amdgcn_wmma_f32_16x16x32_bf16(
                false, as_v16bf(am), false, B, (short)0, acc[l], false, false);
        }
    }
#pragma unroll
    for (int l = 0; l < NLAB; ++l)
#pragma unroll
        for (int r = 0; r < 8; ++r) {
            int row = itile * 16 + r + half * 8;
            int col = jtile * 16 + (lane & 15);
            atomicAdd(&cov[l * 65536 + row * 256 + col], acc[l][r]);
        }
}

__global__ void __launch_bounds__(256)
k_apply_slow(const float* __restrict__ x, const int* __restrict__ seg,
             const float* __restrict__ ws, float* __restrict__ out) {
    __shared__ float muc[NLAB * CCH];
    __shared__ float mus[NLAB * CCH];
    __shared__ float gd[NLAB];
    for (int i = threadIdx.x; i < NLAB * CCH; i += blockDim.x) {
        muc[i] = ws[OFF_MU_C + i];
        mus[i] = ws[OFF_MU_S + i];
    }
    if (threadIdx.x < NLAB) gd[threadIdx.x] = ws[OFF_GUIDE + threadIdx.x];
    __syncthreads();

    const float* T = ws + OFF_T;
    const int wave  = threadIdx.x >> 5;
    const int lane  = threadIdx.x & 31;
    const int itile = blockIdx.y * 8 + wave;
    const int px0   = blockIdx.x * 16;
    const int n = lane & 15, half = lane >> 4;
    const int px = px0 + n;
    const int labn = seg[px];
    const float gn = gd[labn];
    const int arow = itile * 16 + (lane & 15);

    const v8f vzero = {0.f, 0.f, 0.f, 0.f, 0.f, 0.f, 0.f, 0.f};
    v8f acc = vzero;
    for (int c0 = 0; c0 < CCH; c0 += 32) {
        uint32_t bU[8];
#pragma unroll
        for (int v = 0; v < 8; ++v) {
            int k = c0 + ((v >> 2) << 4) + (half << 3) + ((v & 3) << 1);
            float b0 = x[(size_t)k * NPIX + px]       - muc[labn * CCH + k];
            float b1 = x[(size_t)(k + 1) * NPIX + px] - muc[labn * CCH + k + 1];
            bU[v] = pack2bf(b0, b1);
        }
#pragma unroll
        for (int l = 0; l < NLAB; ++l) {
            if (__any(labn == l)) {
                uint32_t aU[8], bm[8];
                const float* Tl = T + l * 65536 + arow * CCH;
#pragma unroll
                for (int v = 0; v < 8; ++v) {
                    int k = c0 + ((v >> 2) << 4) + (half << 3) + ((v & 3) << 1);
                    aU[v] = pack2bf(Tl[k], Tl[k + 1]);
                    bm[v] = (labn == l) ? bU[v] : 0u;
                }
                acc = __builtin_amdgcn_wmma_f32_16x16x32_bf16(
                    false, as_v16bf(aU), false, as_v16bf(bm),
                    (short)0, acc, false, false);
            }
        }
    }
#pragma unroll
    for (int r = 0; r < 8; ++r) {
        int c = itile * 16 + r + half * 8;
        float val = (gn > 0.5f) ? (acc[r] + mus[labn * CCH + c])
                                : x[(size_t)c * NPIX + px];
        out[(size_t)c * NPIX + px] = val;
    }
}

// ---------------------------------------------------------------------------
extern "C" void kernel_launch(void* const* d_in, const int* in_sizes, int n_in,
                              void* d_out, int out_size, void* d_ws, size_t ws_size,
                              hipStream_t stream) {
    (void)in_sizes; (void)n_in; (void)out_size;
    const float* xc   = (const float*)d_in[0];
    const float* xs   = (const float*)d_in[1];
    const int*   segc = (const int*)d_in[2];
    const int*   segs = (const int*)d_in[3];
    float* out = (float*)d_out;
    float* ws  = (float*)d_ws;
    uint32_t* ws32 = (uint32_t*)d_ws;
    const bool fast = ws_size >= FAST_BYTES;

    k_zero<<<1024, 256, 0, stream>>>(ws, ZERO_FLOATS);
    k_hist<<<64, 256, 0, stream>>>(segc, ws + OFF_CNT_C);
    k_hist<<<64, 256, 0, stream>>>(segs, ws + OFF_CNT_S);
    k_sums<<<dim3(CCH, 16), 256, 0, stream>>>(xc, segc, ws + OFF_SUM_C);
    k_sums<<<dim3(CCH, 16), 256, 0, stream>>>(xs, segs, ws + OFF_SUM_S);
    k_means<<<NLAB, CCH, 0, stream>>>(ws);

    if (fast) {
        k_packx<<<4096, 256, 0, stream>>>(xc, ws32 + OFF_XBF_C);
        k_packx<<<4096, 256, 0, stream>>>(xs, ws32 + OFF_XBF_S);
        k_masks<<<1024, 256, 0, stream>>>(segc, ws32 + OFF_MSK_C);
        k_masks<<<1024, 256, 0, stream>>>(segs, ws32 + OFF_MSK_S);
        k_cov_fast<<<dim3(256, 16, 2), 32, 0, stream>>>(ws);
    } else {
        k_cov_slow<<<dim3(256, 16, 2), 32, 0, stream>>>(xc, xs, segc, segs, ws);
    }

    k_covfin<<<dim3(NLAB * CCH, 1, 2), CCH, 0, stream>>>(ws);
    k_chol<<<16, CCH, 0, stream>>>(ws);
    k_trinv<<<NLAB, CCH, 0, stream>>>(ws);
    k_formT<<<dim3(NLAB, CCH), CCH, 0, stream>>>(ws);

    if (fast) {
        k_packT<<<256, 256, 0, stream>>>(ws + OFF_T, ws32 + OFF_TBF);
        k_xtc<<<NPIX / 256, 256, 0, stream>>>(xc, segc, ws, ws32 + OFF_XTC);
        k_apply_fast<<<dim3(NPIX / 16, 2), 256, 0, stream>>>(xc, segc, ws, out);
    } else {
        k_apply_slow<<<dim3(NPIX / 16, 2), 256, 0, stream>>>(xc, segc, ws, out);
    }
}
